// TransformerEncoder_80058190397606
// MI455X (gfx1250) — matrix operand order repaired and checked by reading, not compiled, not run on hardware
//
#include <hip/hip_runtime.h>
#include <hip/hip_bf16.h>

// ---------------------------------------------------------------------------
// MI455X (gfx1250) implementation. wave32, WMMA f32<-f16 16x16x32.
// Batched WMMA GEMM: block tile 128x64, BK=32, 8 waves, each wave 32x32
// (2 A-frags x 2 B-frags = 4 v_wmma per k-step). Double-buffered LDS with
// GLOBAL_LOAD_ASYNC_TO_LDS_B128 staging (ASYNCcnt) for contiguous paths;
// transposed paths use 128-bit global loads + LDS-side transpose.
// ---------------------------------------------------------------------------

typedef _Float16 h16;
typedef __attribute__((ext_vector_type(16))) _Float16 v16h;
typedef __attribute__((ext_vector_type(8)))  _Float16 v8h;
typedef __attribute__((ext_vector_type(8)))  float    v8f;

#if defined(__HIP_DEVICE_COMPILE__) && __has_builtin(__builtin_amdgcn_global_load_async_to_lds_b128)
#define ASYNC_LDS 1
typedef int v4i __attribute__((vector_size(16)));
typedef __attribute__((address_space(1))) void gas1;     // global void
typedef __attribute__((address_space(3))) void las3;     // LDS void
typedef __attribute__((address_space(1))) v4i gv4i;      // global v4i
typedef __attribute__((address_space(3))) v4i lv4i;      // LDS v4i
#define GPTR(p) ((gv4i*)(gas1*)(void*)(p))
#define LPTR(p) ((lv4i*)(las3*)(void*)(p))
#else
#define ASYNC_LDS 0
#endif

__device__ __forceinline__ void wait_async_lds() {
#if ASYNC_LDS
#if __has_builtin(__builtin_amdgcn_s_wait_asynccnt)
    __builtin_amdgcn_s_wait_asynccnt(0);
#else
    asm volatile("s_wait_asynccnt 0x0" ::: "memory");
#endif
#endif
}

#define B_   64
#define T_   256
#define R_   256
#define H_   256
#define NH_  8
#define DH_  32
#define FF_  512
#define LAT_ 64
#define L_   2
#define BT_  (B_ * T_)

// ---------------------------------------------------------------------------
// C[z] = alpha * A[z] @ B[z] (+bias), epilogue: relu / clamp+nan0, f32/f16 out.
//  A: [M,K] f16 (lda); transA => stored [K,M].
//  B: [K,N] f16 (ldb); transB => stored [N,K].
//  z = blockIdx.z; z1 = z/zdiv, z2 = z%zdiv; ptr += z1*s?1 + z2*s?2.
//  Requires K % 32 == 0 and all dims % 8 == 0 (true for this model).
// ---------------------------------------------------------------------------
__global__ __launch_bounds__(256) void wmma_gemm_kernel(
    const h16* __restrict__ A, const h16* __restrict__ Bm,
    const float* __restrict__ bias,
    float* __restrict__ Cf, h16* __restrict__ Ch,
    int M, int N, int K, int lda, int ldb, int ldc,
    long sA1, long sA2, long sB1, long sB2, long sC1, long sC2,
    int zdiv, int transA, int transB, float alpha, int relu, int clampNan)
{
    const int z  = blockIdx.z;
    const int z1 = z / zdiv, z2 = z % zdiv;
    A  += (long)z1 * sA1 + (long)z2 * sA2;
    Bm += (long)z1 * sB1 + (long)z2 * sB2;
    if (Cf) Cf += (long)z1 * sC1 + (long)z2 * sC2;
    if (Ch) Ch += (long)z1 * sC1 + (long)z2 * sC2;

    __shared__ alignas(128) h16 As[2][128][32];  // [buf][m][k]
    __shared__ alignas(128) h16 Bs[2][64][32];   // [buf][n][k] (K-transposed)

    const int tid  = threadIdx.x;
    const int lane = tid & 31;
    const int w    = tid >> 5;
    const int wm   = w & 3;          // 4 waves along M (32 rows each)
    const int wn   = w >> 2;         // 2 waves along N (32 cols each)
    const int l16  = lane & 15;
    const int hs   = lane >> 4;      // wave32 lane half-select
    const int bm   = blockIdx.y * 128;
    const int bn   = blockIdx.x * 64;

    v8h ra0, ra1, rb0;               // register staging (transposed paths /
                                     // async fallback)

    // Issue stage for k-block kb into buffer `buf`: async direct-to-LDS for
    // contiguous layouts, 128-bit global loads into registers otherwise.
    auto stageIssue = [&](int kb, int buf) {
        if (!transA) {               // A[m][k]: 8 contiguous k per thread
            int row = tid >> 2, c0 = (tid & 3) << 3;
#if ASYNC_LDS
            if (bm + row < M)
                __builtin_amdgcn_global_load_async_to_lds_b128(
                    GPTR(A + (long)(bm + row) * lda + kb + c0),
                    LPTR(&As[buf][row][c0]), 0, 0);
            if (bm + row + 64 < M)
                __builtin_amdgcn_global_load_async_to_lds_b128(
                    GPTR(A + (long)(bm + row + 64) * lda + kb + c0),
                    LPTR(&As[buf][row + 64][c0]), 0, 0);
#else
            ra0 = {}; ra1 = {};
            if (bm + row < M)
                ra0 = *(const v8h*)(A + (long)(bm + row) * lda + kb + c0);
            if (bm + row + 64 < M)
                ra1 = *(const v8h*)(A + (long)(bm + row + 64) * lda + kb + c0);
#endif
        } else {                     // A[k][m]: 8 contiguous m per thread
            int k = tid & 31, m0 = (tid >> 5) << 3;
            const h16* ap = A + (long)(kb + k) * lda + bm;
            ra0 = {}; ra1 = {};
            if (bm + m0 < M)      ra0 = *(const v8h*)(ap + m0);
            if (bm + m0 + 64 < M) ra1 = *(const v8h*)(ap + m0 + 64);
        }
        if (transB) {                // B[n][k]: 8 contiguous k per thread
            int n = tid >> 2, c0 = (tid & 3) << 3;
#if ASYNC_LDS
            if (bn + n < N)
                __builtin_amdgcn_global_load_async_to_lds_b128(
                    GPTR(Bm + (long)(bn + n) * ldb + kb + c0),
                    LPTR(&Bs[buf][n][c0]), 0, 0);
#else
            rb0 = {};
            if (bn + n < N)
                rb0 = *(const v8h*)(Bm + (long)(bn + n) * ldb + kb + c0);
#endif
        } else {                     // B[k][n]: 8 contiguous n per thread
            int k = tid & 31, n0 = (tid >> 5) << 3;
            rb0 = {};
            if (bn + n0 < N)
                rb0 = *(const v8h*)(Bm + (long)(kb + k) * ldb + bn + n0);
        }
    };
    // Store register-staged data into LDS buffer `buf` (transpose on write).
    auto stageStore = [&](int buf) {
        if (!transA) {
#if !ASYNC_LDS
            int row = tid >> 2, c0 = (tid & 3) << 3;
            *(v8h*)&As[buf][row][c0]      = ra0;
            *(v8h*)&As[buf][row + 64][c0] = ra1;
#endif
        } else {
            int k = tid & 31, m0 = (tid >> 5) << 3;
            #pragma unroll
            for (int i = 0; i < 8; ++i) {
                As[buf][m0 + i][k]      = ra0[i];
                As[buf][m0 + 64 + i][k] = ra1[i];
            }
        }
        if (transB) {
#if !ASYNC_LDS
            int n = tid >> 2, c0 = (tid & 3) << 3;
            *(v8h*)&Bs[buf][n][c0] = rb0;
#endif
        } else {
            int k = tid & 31, n0 = (tid >> 5) << 3;
            #pragma unroll
            for (int i = 0; i < 8; ++i) Bs[buf][n0 + i][k] = rb0[i];
        }
    };

    v8f acc[2][2] = {};

    // prologue: stage tile 0 into buffer 0
    stageIssue(0, 0);
    stageStore(0);
    wait_async_lds();
    __syncthreads();

    int cur = 0;
    for (int kb = 0; kb < K; kb += 32) {
        const bool nxt = (kb + 32) < K;
        if (nxt) stageIssue(kb + 32, cur ^ 1);   // overlaps the WMMAs below

        // A 16x32 frag: lane m = lane%16; elem e -> K = hs*16 + e (contiguous)
        //  => two contiguous ds_load_b128.
        v16h afr[2];
        #pragma unroll
        for (int i = 0; i < 2; ++i) {
            int am = wm * 32 + i * 16 + l16;
            v8h lo = *(const v8h*)&As[cur][am][hs * 16];
            v8h hi = *(const v8h*)&As[cur][am][hs * 16 + 8];
            #pragma unroll
            for (int e = 0; e < 8; ++e) { afr[i][e] = lo[e]; afr[i][8 + e] = hi[e]; }
        }
        // B 32x16 frag: lane n = lane%16; elem e -> K = hs*16 + e (contiguous).
        v16h bfr[2];
        #pragma unroll
        for (int j = 0; j < 2; ++j) {
            int nb = wn * 32 + j * 16 + l16;
            bfr[j] = *(const v16h*)&Bs[cur][nb][hs * 16];
        }
        #pragma unroll
        for (int i = 0; i < 2; ++i)
            #pragma unroll
            for (int j = 0; j < 2; ++j)
                acc[i][j] = __builtin_amdgcn_wmma_f32_16x16x32_f16(
                    false, afr[i], false, bfr[j], (short)0, acc[i][j], false, false);

        if (nxt) {
            stageStore(cur ^ 1);     // reg-path LDS writes for next tile
            wait_async_lds();        // next buffer's async copies complete
            __syncthreads();         // publish buffer, retire reads of `cur`
        }
        cur ^= 1;
    }

    // ---- epilogue: C 16x16 f32 layout: VGPR r -> M = r + 8*hs, N = lane%16 ----
    #pragma unroll
    for (int i = 0; i < 2; ++i) {
        #pragma unroll
        for (int r = 0; r < 8; ++r) {
            int gm = bm + wm * 32 + i * 16 + hs * 8 + r;
            if (gm >= M) continue;
            #pragma unroll
            for (int j = 0; j < 2; ++j) {
                int gn = bn + wn * 32 + j * 16 + l16;
                if (gn >= N) continue;
                float v = acc[i][j][r] * alpha;
                if (bias) v += bias[gn];
                if (relu) v = v > 0.f ? v : 0.f;
                if (clampNan) {
                    if (v != v) v = 0.f;
                    v = fminf(fmaxf(v, -1.f), 1.f);
                }
                long ci = (long)gm * ldc + gn;
                if (Cf) Cf[ci] = v;
                if (Ch) Ch[ci] = (h16)v;
            }
        }
    }
}

// ---------------------------------------------------------------------------
// Elementwise / reduction kernels (wave32-safe: only __syncthreads used).
// ---------------------------------------------------------------------------
__global__ void cvt_kernel(const float* __restrict__ x, h16* __restrict__ y, long n) {
    long i = (long)blockIdx.x * blockDim.x + threadIdx.x;
    if (i < n) y[i] = (h16)x[i];
}

__global__ void avg_cvt_kernel(const float* __restrict__ a, const float* __restrict__ b,
                               h16* __restrict__ y, long n) {
    long i = (long)blockIdx.x * blockDim.x + threadIdx.x;
    if (i < n) y[i] = (h16)(0.5f * (a[i] + b[i]));
}

// LayerNorm (+optional residual, +optional LeakyReLU(0.1)), one block per row.
__global__ void ln_act_kernel(const float* __restrict__ X, const float* __restrict__ Res,
                              const float* __restrict__ g, const float* __restrict__ bb,
                              float* __restrict__ Yf, h16* __restrict__ Yh,
                              int C, int act)
{
    int row = blockIdx.x, tid = threadIdx.x;
    __shared__ float red[256];
    long base = (long)row * C;
    float v = X[base + tid];
    if (Res) v += Res[base + tid];
    red[tid] = v; __syncthreads();
    for (int s = C >> 1; s > 0; s >>= 1) { if (tid < s) red[tid] += red[tid + s]; __syncthreads(); }
    float mean = red[0] / C; __syncthreads();
    float d = v - mean;
    red[tid] = d * d; __syncthreads();
    for (int s = C >> 1; s > 0; s >>= 1) { if (tid < s) red[tid] += red[tid + s]; __syncthreads(); }
    float var = red[0] / C;
    float y = d * rsqrtf(var + 1e-5f) * g[tid] + bb[tid];
    if (act) y = (y >= 0.f) ? y : 0.1f * y;
    if (Yf) Yf[base + tid] = y;
    if (Yh) Yh[base + tid] = (h16)y;
}

// Row softmax over 256 columns; f32 in, f16 out.
__global__ void softmax_kernel(const float* __restrict__ S, h16* __restrict__ P) {
    int row = blockIdx.x, tid = threadIdx.x;
    __shared__ float red[256];
    long base = (long)row * 256;
    float v = S[base + tid];
    red[tid] = v; __syncthreads();
    for (int s = 128; s > 0; s >>= 1) { if (tid < s) red[tid] = fmaxf(red[tid], red[tid + s]); __syncthreads(); }
    float mx = red[0]; __syncthreads();
    float e = __expf(v - mx);
    red[tid] = e; __syncthreads();
    for (int s = 128; s > 0; s >>= 1) { if (tid < s) red[tid] += red[tid + s]; __syncthreads(); }
    P[base + tid] = (h16)(e / red[0]);
}

// Pearson prep: one block per (b,r); center over t, divide by unbiased std.
// Y: [B,T,R] f32 -> CN: [B,R,T] f16 (transposed, normalized).
__global__ void corrprep_kernel(const float* __restrict__ Y, h16* __restrict__ CN) {
    int br = blockIdx.x;
    int b = br >> 8, r = br & 255;
    int tid = threadIdx.x;     // = t
    __shared__ float red[256];
    float v = Y[(long)b * T_ * R_ + (long)tid * R_ + r];
    red[tid] = v; __syncthreads();
    for (int s = 128; s > 0; s >>= 1) { if (tid < s) red[tid] += red[tid + s]; __syncthreads(); }
    float mean = red[0] / T_; __syncthreads();
    float c = v - mean;
    red[tid] = c * c; __syncthreads();
    for (int s = 128; s > 0; s >>= 1) { if (tid < s) red[tid] += red[tid + s]; __syncthreads(); }
    float stdv = sqrtf(red[0] / (float)(T_ - 1));
    CN[(long)b * R_ * T_ + (long)r * T_ + tid] = (h16)(c / (stdv + 1e-8f));
}

// pooled[b,c] = mean_t h[b,t,c]; writes f32 + f16.
__global__ void pool_kernel(const float* __restrict__ Hf,
                            float* __restrict__ Pf, h16* __restrict__ Ph) {
    int idx = blockIdx.x * 256 + threadIdx.x;   // b*H + c
    int b = idx >> 8, c = idx & 255;
    float s = 0.f;
    for (int t = 0; t < T_; ++t) s += Hf[(long)b * T_ * H_ + (long)t * H_ + c];
    s *= (1.f / T_);
    Pf[idx] = s; Ph[idx] = (h16)s;
}

// ---------------------------------------------------------------------------
// Host-side helpers
// ---------------------------------------------------------------------------
static void gemm_launch(hipStream_t st, const h16* A, const h16* Bm, const float* bias,
                        float* Cf, h16* Ch, int M, int N, int K,
                        int lda, int ldb, int ldc,
                        int batch, int zdiv,
                        long sA1, long sA2, long sB1, long sB2, long sC1, long sC2,
                        int transA, int transB, float alpha, int relu, int clampNan)
{
    dim3 grid((unsigned)((N + 63) / 64), (unsigned)((M + 127) / 128), (unsigned)batch);
    wmma_gemm_kernel<<<grid, dim3(256), 0, st>>>(
        A, Bm, bias, Cf, Ch, M, N, K, lda, ldb, ldc,
        sA1, sA2, sB1, sB2, sC1, sC2, zdiv, transA, transB, alpha, relu, clampNan);
}

static void cvt(hipStream_t st, const float* x, h16* y, long n) {
    cvt_kernel<<<(unsigned)((n + 255) / 256), 256, 0, st>>>(x, y, n);
}

// ---------------------------------------------------------------------------
// Workspace map (~292 MB; lifetimes reused):
// ---------------------------------------------------------------------------
#define MB_ (1ull << 20)
#define OFF_HF    (0 * MB_)          // float [BT,H]      16MB (persistent)
#define OFF_HH    (16 * MB_)         // f16   [BT,H]       8MB (persistent)
#define OFF_TMPF  (24 * MB_)         // float [BT,FF max] 32MB (pre-LN GEMM outs)
#define OFF_QKVH  (56 * MB_)         // f16   [BT,3H]     24MB (per layer)
#define OFF_AVGH  (56 * MB_)         // f16   [B,R,R]      8MB (pre-stage, dead before qkv)
#define OFF_XH    (64 * MB_)         // f16   [B,T,R]      8MB (pre-stage)
#define OFF_H1H   (72 * MB_)         // f16   [BT,R]       8MB (pre-stage)
#define OFF_SCORE (80 * MB_)         // float [B,NH,T,T] 128MB (per layer)
#define OFF_YF    (80 * MB_)         // float [BT,R]      16MB (post-loop, reuses score)
#define OFF_CNH   (96 * MB_)         // f16   [B,R,T]      8MB (post-loop)
#define OFF_PH    (208 * MB_)        // f16   [B,NH,T,T]  64MB (per layer)
#define OFF_OH    (272 * MB_)        // f16   [BT,H]       8MB (dead before ffn1)
#define OFF_FFH   (272 * MB_)        // f16   [BT,FF]     16MB
#define OFF_POOLF (288 * MB_)        // float [B,H]
#define OFF_POOLH (288 * MB_ + 64 * 1024)
#define OFF_T1F   (288 * MB_ + 96 * 1024)
#define OFF_T1H   (288 * MB_ + 128 * 1024)
#define OFF_W     (289 * MB_)        // f16 weight pool (~2.5MB)
// f16 weight pool sub-offsets (in halves):
#define WO_FPW   0
#define WO_QKVW  (WO_FPW + H_ * R_)               //  65536
#define WO_AOW   (WO_QKVW + L_ * 3 * H_ * H_)     // +393216
#define WO_FF1W  (WO_AOW + L_ * H_ * H_)          // +131072
#define WO_FF2W  (WO_FF1W + L_ * FF_ * H_)        // +262144
#define WO_OUTW  (WO_FF2W + L_ * H_ * FF_)        // +262144
#define WO_MUW1  (WO_OUTW + R_ * H_)              // + 65536
#define WO_MUW2  (WO_MUW1 + (H_ / 2) * H_)        // + 32768
#define WO_LVW1  (WO_MUW2 + LAT_ * (H_ / 2))      // +  8192
#define WO_LVW2  (WO_LVW1 + (H_ / 2) * H_)        // + 32768

extern "C" void kernel_launch(void* const* d_in, const int* in_sizes, int n_in,
                              void* d_out, int out_size, void* d_ws, size_t ws_size,
                              hipStream_t stream)
{
    (void)in_sizes; (void)n_in; (void)out_size; (void)ws_size;
    // ---- inputs (setup_inputs order) ----
    const float* x         = (const float*)d_in[0];
    const float* node_view = (const float*)d_in[1];
    const float* edge_view = (const float*)d_in[2];
    const float* fp_w      = (const float*)d_in[3];
    const float* fp_b      = (const float*)d_in[4];
    const float* fp_g      = (const float*)d_in[5];
    const float* fp_beta   = (const float*)d_in[6];
    const float* qkv_w     = (const float*)d_in[7];
    const float* qkv_b     = (const float*)d_in[8];
    const float* aow       = (const float*)d_in[9];
    const float* aob       = (const float*)d_in[10];
    const float* ln1_g     = (const float*)d_in[11];
    const float* ln1_b     = (const float*)d_in[12];
    const float* ff1_w     = (const float*)d_in[13];
    const float* ff1_b     = (const float*)d_in[14];
    const float* ff2_w     = (const float*)d_in[15];
    const float* ff2_b     = (const float*)d_in[16];
    const float* ln2_g     = (const float*)d_in[17];
    const float* ln2_b     = (const float*)d_in[18];
    const float* out_w     = (const float*)d_in[19];
    const float* out_b     = (const float*)d_in[20];
    const float* mu_w1     = (const float*)d_in[21];
    const float* mu_b1     = (const float*)d_in[22];
    const float* mu_g      = (const float*)d_in[23];
    const float* mu_beta   = (const float*)d_in[24];
    const float* mu_w2     = (const float*)d_in[25];
    const float* mu_b2     = (const float*)d_in[26];
    const float* lv_w1     = (const float*)d_in[27];
    const float* lv_b1     = (const float*)d_in[28];
    const float* lv_g      = (const float*)d_in[29];
    const float* lv_beta   = (const float*)d_in[30];
    const float* lv_w2     = (const float*)d_in[31];
    const float* lv_b2     = (const float*)d_in[32];

    char* ws = (char*)d_ws;
    float* hF    = (float*)(ws + OFF_HF);
    h16*   hH    = (h16*)  (ws + OFF_HH);
    float* tmpF  = (float*)(ws + OFF_TMPF);
    h16*   qkvH  = (h16*)  (ws + OFF_QKVH);
    h16*   avgH  = (h16*)  (ws + OFF_AVGH);
    h16*   xH    = (h16*)  (ws + OFF_XH);
    h16*   h1H   = (h16*)  (ws + OFF_H1H);
    float* scoreF= (float*)(ws + OFF_SCORE);
    float* yF    = (float*)(ws + OFF_YF);
    h16*   cnH   = (h16*)  (ws + OFF_CNH);
    h16*   pH    = (h16*)  (ws + OFF_PH);
    h16*   oH    = (h16*)  (ws + OFF_OH);
    h16*   ffH   = (h16*)  (ws + OFF_FFH);
    float* poolF = (float*)(ws + OFF_POOLF);
    h16*   poolH = (h16*)  (ws + OFF_POOLH);
    float* t1F   = (float*)(ws + OFF_T1F);
    h16*   t1H   = (h16*)  (ws + OFF_T1H);
    h16*   wp    = (h16*)  (ws + OFF_W);

    float* outMu   = (float*)d_out;             // [64,64]
    float* outLv   = outMu + B_ * LAT_;         // [64,64]
    float* outCorr = outLv + B_ * LAT_;         // [64,256,256]

    // ---- stage 0: f32 -> f16 conversions (weights once, activations) ----
    cvt(stream, x, xH, (long)BT_ * R_);
    avg_cvt_kernel<<<(unsigned)(((long)B_ * R_ * R_ + 255) / 256), 256, 0, stream>>>(
        node_view, edge_view, avgH, (long)B_ * R_ * R_);
    cvt(stream, fp_w,  wp + WO_FPW,  H_ * R_);
    cvt(stream, qkv_w, wp + WO_QKVW, (long)L_ * 3 * H_ * H_);
    cvt(stream, aow,   wp + WO_AOW,  (long)L_ * H_ * H_);
    cvt(stream, ff1_w, wp + WO_FF1W, (long)L_ * FF_ * H_);
    cvt(stream, ff2_w, wp + WO_FF2W, (long)L_ * H_ * FF_);
    cvt(stream, out_w, wp + WO_OUTW, R_ * H_);
    cvt(stream, mu_w1, wp + WO_MUW1, (H_ / 2) * H_);
    cvt(stream, mu_w2, wp + WO_MUW2, LAT_ * (H_ / 2));
    cvt(stream, lv_w1, wp + WO_LVW1, (H_ / 2) * H_);
    cvt(stream, lv_w2, wp + WO_LVW2, LAT_ * (H_ / 2));

    // ---- message passing: h1[b] = avg[b]^T (T x T) @ x[b] (T x R) ----
    gemm_launch(stream, avgH, xH, nullptr, nullptr, h1H,
                T_, R_, T_, R_, R_, R_,
                B_, 1, (long)R_ * R_, 0, (long)T_ * R_, 0, (long)T_ * R_, 0,
                /*transA*/1, /*transB*/0, 1.f, 0, 0);

    // ---- feature_proj: Linear -> LN -> LeakyReLU ----
    gemm_launch(stream, h1H, wp + WO_FPW, fp_b, tmpF, nullptr,
                BT_, H_, R_, R_, R_, H_,
                1, 1, 0, 0, 0, 0, 0, 0, 0, /*transB*/1, 1.f, 0, 0);
    ln_act_kernel<<<BT_, H_, 0, stream>>>(tmpF, nullptr, fp_g, fp_beta, hF, hH, H_, 1);

    // ---- transformer layers ----
    const float scale = 0.17677669529663687f;   // 1/sqrt(32)
    for (int i = 0; i < L_; ++i) {
        // qkv = h @ qkv_w^T + b  -> f16 [BT, 3H]
        gemm_launch(stream, hH, wp + WO_QKVW + (long)i * 3 * H_ * H_, qkv_b + i * 3 * H_,
                    nullptr, qkvH, BT_, 3 * H_, H_, H_, H_, 3 * H_,
                    1, 1, 0, 0, 0, 0, 0, 0, 0, 1, 1.f, 0, 0);
        // scores[b,hd] = scale * q @ k^T   (batch = B*NH, zdiv = NH)
        gemm_launch(stream, qkvH, qkvH + H_, nullptr, scoreF, nullptr,
                    T_, T_, DH_, 3 * H_, 3 * H_, T_,
                    B_ * NH_, NH_,
                    (long)T_ * 3 * H_, DH_, (long)T_ * 3 * H_, DH_,
                    (long)NH_ * T_ * T_, (long)T_ * T_,
                    0, /*transB*/1, scale, 0, 0);
        softmax_kernel<<<B_ * NH_ * T_, T_, 0, stream>>>(scoreF, pH);
        // o[b,hd] = P @ v  -> f16, written into [BT, H] at column hd*32
        gemm_launch(stream, pH, qkvH + 2 * H_, nullptr, nullptr, oH,
                    T_, DH_, T_, T_, 3 * H_, H_,
                    B_ * NH_, NH_,
                    (long)NH_ * T_ * T_, (long)T_ * T_,
                    (long)T_ * 3 * H_, DH_,
                    (long)T_ * H_, DH_,
                    0, /*transB*/0, 1.f, 0, 0);
        // attn_out projection -> f32, then residual + LN1
        gemm_launch(stream, oH, wp + WO_AOW + (long)i * H_ * H_, aob + i * H_,
                    tmpF, nullptr, BT_, H_, H_, H_, H_, H_,
                    1, 1, 0, 0, 0, 0, 0, 0, 0, 1, 1.f, 0, 0);
        ln_act_kernel<<<BT_, H_, 0, stream>>>(tmpF, hF, ln1_g + i * H_, ln1_b + i * H_,
                                              hF, hH, H_, 0);
        // FFN: relu(h @ ff1^T + b1) @ ff2^T + b2, residual + LN2
        gemm_launch(stream, hH, wp + WO_FF1W + (long)i * FF_ * H_, ff1_b + i * FF_,
                    nullptr, ffH, BT_, FF_, H_, H_, H_, FF_,
                    1, 1, 0, 0, 0, 0, 0, 0, 0, 1, 1.f, /*relu*/1, 0);
        gemm_launch(stream, ffH, wp + WO_FF2W + (long)i * H_ * FF_, ff2_b + i * H_,
                    tmpF, nullptr, BT_, H_, FF_, FF_, FF_, H_,
                    1, 1, 0, 0, 0, 0, 0, 0, 0, 1, 1.f, 0, 0);
        ln_act_kernel<<<BT_, H_, 0, stream>>>(tmpF, hF, ln2_g + i * H_, ln2_b + i * H_,
                                              hF, hH, H_, 0);
    }

    // ---- output linear + Pearson correlation ----
    gemm_launch(stream, hH, wp + WO_OUTW, out_b, yF, nullptr,
                BT_, R_, H_, H_, H_, R_,
                1, 1, 0, 0, 0, 0, 0, 0, 0, 1, 1.f, 0, 0);
    corrprep_kernel<<<B_ * R_, T_, 0, stream>>>(yF, cnH);
    // corr[b] = (1/(T-1)) * cn[b] @ cn[b]^T, clamp [-1,1], NaN->0 -> d_out
    gemm_launch(stream, cnH, cnH, nullptr, outCorr, nullptr,
                R_, R_, T_, T_, T_, R_,
                B_, 1, (long)R_ * T_, 0, (long)R_ * T_, 0, (long)R_ * R_, 0,
                0, /*transB*/1, 1.f / (float)(T_ - 1), 0, /*clampNan*/1);

    // ---- heads ----
    pool_kernel<<<(B_ * H_) / 256, 256, 0, stream>>>(hF, poolF, poolH);
    // mu
    gemm_launch(stream, poolH, wp + WO_MUW1, mu_b1, t1F, nullptr,
                B_, H_ / 2, H_, H_, H_, H_ / 2,
                1, 1, 0, 0, 0, 0, 0, 0, 0, 1, 1.f, 0, 0);
    ln_act_kernel<<<B_, H_ / 2, 0, stream>>>(t1F, nullptr, mu_g, mu_beta,
                                             nullptr, t1H, H_ / 2, 1);
    gemm_launch(stream, t1H, wp + WO_MUW2, mu_b2, outMu, nullptr,
                B_, LAT_, H_ / 2, H_ / 2, H_ / 2, LAT_,
                1, 1, 0, 0, 0, 0, 0, 0, 0, 1, 1.f, 0, 0);
    // logvar
    gemm_launch(stream, poolH, wp + WO_LVW1, lv_b1, t1F, nullptr,
                B_, H_ / 2, H_, H_, H_, H_ / 2,
                1, 1, 0, 0, 0, 0, 0, 0, 0, 1, 1.f, 0, 0);
    ln_act_kernel<<<B_, H_ / 2, 0, stream>>>(t1F, nullptr, lv_g, lv_beta,
                                             nullptr, t1H, H_ / 2, 1);
    gemm_launch(stream, t1H, wp + WO_LVW2, lv_b2, outLv, nullptr,
                B_, LAT_, H_ / 2, H_ / 2, H_ / 2, LAT_,
                1, 1, 0, 0, 0, 0, 0, 0, 0, 1, 1.f, 0, 0);
}
